// PointNetSetAbstraction_4080218931821
// MI455X (gfx1250) — matrix operand order repaired and checked by reading, not compiled, not run on hardware
//
#include <hip/hip_runtime.h>

typedef _Float16 h8  __attribute__((ext_vector_type(8)));
typedef _Float16 f16x16 __attribute__((ext_vector_type(16)));
typedef float    f32x8  __attribute__((ext_vector_type(8)));

// ---------------- problem constants ----------------
constexpr int BB   = 8;
constexpr int NPT  = 8192;
constexpr int SS   = 2048;
constexpr int KK   = 32;
constexpr int CPTS = 64;
constexpr float RAD2 = 0.25f * 0.25f;
constexpr int C0   = 3 + CPTS;   // 67
constexpr int C0P  = 96;         // padded to 3 x 32 for WMMA K-chunks
constexpr long long MTOT = (long long)BB * SS * KK;   // 524288 rows
constexpr int TILEM = 128;       // rows per block (8 waves x 16)

// stats layout per layer (floats): [sum 128][sumsq 128][scale 128][shift 128]
constexpr int STATS_STRIDE = 512;

// ---------------- CDNA5 async global->LDS copy (ASYNCcnt path) ----------------
__device__ __forceinline__ void async_copy_b128(unsigned lds_addr, const void* gaddr) {
    // ISA 15.18.3 opcode 98: GLOBAL_LOAD_ASYNC_TO_LDS_B128  vdst(LDS), vaddr(64b), off
    asm volatile("global_load_async_to_lds_b128 %0, %1, off"
                 :: "v"(lds_addr), "v"(gaddr)
                 : "memory");
}
__device__ __forceinline__ void wait_async_zero() {
    asm volatile("s_wait_asynccnt 0x0" ::: "memory");
}
__device__ __forceinline__ unsigned lds_addr_of(const void* p) {
    // low 32 bits of a flat shared-memory address are the LDS byte address (ISA 10.2)
    return (unsigned)(unsigned long long)p;
}

// ---------------- WMMA fragment loaders (ISA 7.12.2 layouts) ----------------
__device__ __forceinline__ f16x16 load_frag_a(const _Float16* row, int kb, int hi4) {
    // lane<16: K = kb+0..7 and kb+16..23 ; lane>=16: K = kb+8..15 and kb+24..31
    h8 lo = *(const h8*)(row + kb + 8 * hi4);
    h8 hi = *(const h8*)(row + kb + 16 + 8 * hi4);
    return __builtin_shufflevector(lo, hi, 0,1,2,3,4,5,6,7,8,9,10,11,12,13,14,15);
}
__device__ __forceinline__ f16x16 load_frag_b(const _Float16* wrow, int kb, int hi4) {
    // lane<16: K = kb+0..7 and kb+16..23 ; lane>=16: K = kb+8..15 and kb+24..31  (B[k][n] = w[n][k])
    h8 lo = *(const h8*)(wrow + kb + 8 * hi4);
    h8 hi = *(const h8*)(wrow + kb + 16 + 8 * hi4);
    return __builtin_shufflevector(lo, hi, 0,1,2,3,4,5,6,7,8,9,10,11,12,13,14,15);
}

// ---------------- 0a) zero BN accumulators ----------------
__global__ void zero_stats_kernel(float* stats) {
    for (int i = threadIdx.x; i < 3 * STATS_STRIDE; i += blockDim.x) stats[i] = 0.f;
}

// ---------------- 0b) one-time f32 -> padded f16 weight conversion ----------
__global__ void prep_weights_kernel(const float* __restrict__ w0,
                                    const float* __restrict__ w1,
                                    const float* __restrict__ w2,
                                    _Float16* __restrict__ w0h,
                                    _Float16* __restrict__ w1h,
                                    _Float16* __restrict__ w2h) {
    const int stride = gridDim.x * blockDim.x;
    const int t = blockIdx.x * blockDim.x + threadIdx.x;
    for (int i = t; i < 64 * C0P; i += stride) {     // [64][96], zero padded
        int n = i / C0P, c = i - n * C0P;
        w0h[i] = (c < C0) ? (_Float16)w0[n * C0 + c] : (_Float16)0.f;
    }
    for (int i = t; i < 64 * 64; i += stride)  w1h[i] = (_Float16)w1[i];
    for (int i = t; i < 128 * 64; i += stride) w2h[i] = (_Float16)w2[i];
}

// ---------------- 1) furthest point sampling ----------------
// one block per batch; 1024 threads; 8 points per thread held in registers
__global__ void __launch_bounds__(1024) fps_kernel(const float* __restrict__ xyz,
                                                   int* __restrict__ fps_idx,
                                                   float* __restrict__ new_xyz) {
    const int b = blockIdx.x;
    const int t = threadIdx.x;
    const float* base = xyz + (long long)b * NPT * 3;

    float px[8], py[8], pz[8], mind[8];
#pragma unroll
    for (int j = 0; j < 8; ++j) {
        int p = t * 8 + j;
        px[j] = base[p * 3 + 0];
        py[j] = base[p * 3 + 1];
        pz[j] = base[p * 3 + 2];
        mind[j] = 1e10f;
    }

    __shared__ float bcx, bcy, bcz;
    __shared__ int   bcur;
    __shared__ float wv[32];
    __shared__ int   wi[32];
    if (t == 0) { bcx = base[0]; bcy = base[1]; bcz = base[2]; bcur = 0; }
    __syncthreads();

    for (int step = 0; step < SS; ++step) {
        if (t == 0) {
            fps_idx[b * SS + step] = bcur;
            float* o = new_xyz + ((long long)b * SS + step) * 3;
            o[0] = bcx; o[1] = bcy; o[2] = bcz;
        }
        const float cx = bcx, cy = bcy, cz = bcz;
        float bestv = -1.f; int besti = 0;
#pragma unroll
        for (int j = 0; j < 8; ++j) {
            float dx = px[j] - cx, dy = py[j] - cy, dz = pz[j] - cz;
            float d = dx * dx + dy * dy + dz * dz;
            mind[j] = fminf(mind[j], d);
            if (mind[j] > bestv) { bestv = mind[j]; besti = t * 8 + j; }
        }
        // wave32 reduce (argmax, smaller index wins ties)
#pragma unroll
        for (int off = 16; off > 0; off >>= 1) {
            float ov = __shfl_down(bestv, off, 32);
            int   oi = __shfl_down(besti, off, 32);
            if (ov > bestv || (ov == bestv && oi < besti)) { bestv = ov; besti = oi; }
        }
        if ((t & 31) == 0) { wv[t >> 5] = bestv; wi[t >> 5] = besti; }
        __syncthreads();
        if (t < 32) {
            bestv = wv[t]; besti = wi[t];
#pragma unroll
            for (int off = 16; off > 0; off >>= 1) {
                float ov = __shfl_down(bestv, off, 32);
                int   oi = __shfl_down(besti, off, 32);
                if (ov > bestv || (ov == bestv && oi < besti)) { bestv = ov; besti = oi; }
            }
            if (t == 0) {
                bcur = besti;
                bcx = base[besti * 3 + 0];
                bcy = base[besti * 3 + 1];
                bcz = base[besti * 3 + 2];
            }
        }
        __syncthreads();
    }
}

// ---------------- 2) ball query: one wave32 per centroid ----------------
__global__ void __launch_bounds__(256) ballq_kernel(const float* __restrict__ xyz,
                                                    const float* __restrict__ new_xyz,
                                                    int* __restrict__ idx) {
    const int gw = (int)((blockIdx.x * (unsigned)blockDim.x + threadIdx.x) >> 5);
    const int lane = threadIdx.x & 31;
    if (gw >= BB * SS) return;
    const int b = gw / SS;
    const float* c = new_xyz + (long long)gw * 3;
    const float cx = c[0], cy = c[1], cz = c[2];
    const float* base = xyz + (long long)b * NPT * 3;
    int* out = idx + (long long)gw * KK;

    int count = 0;
    int first = -1;
    for (int b0 = 0; b0 < NPT && count < KK; b0 += 32) {
        int p = b0 + lane;
        float dx = base[p * 3 + 0] - cx;
        float dy = base[p * 3 + 1] - cy;
        float dz = base[p * 3 + 2] - cz;
        bool hit = (dx * dx + dy * dy + dz * dz) <= RAD2;
        unsigned m = (unsigned)__ballot(hit);
        if (first < 0 && m) first = b0 + __builtin_ctz(m);
        if (hit) {
            int k = count + __popc(m & ((1u << lane) - 1u));
            if (k < KK) out[k] = p;
        }
        count += __popc(m);
    }
    if (first < 0) first = 0;
    int c2 = count < KK ? count : KK;
    for (int k = c2 + lane; k < KK; k += 32) out[k] = first;
}

// ---------------- 3) layer 0: gather + GEMM(67->64) + BN stats ----------------
__global__ void __launch_bounds__(256) gemm0_kernel(const float* __restrict__ xyz,
                                                    const float* __restrict__ points,
                                                    const float* __restrict__ new_xyz,
                                                    const int* __restrict__ idx,
                                                    const _Float16* __restrict__ w0h,
                                                    const float* __restrict__ b0,
                                                    _Float16* __restrict__ x1,
                                                    float* __restrict__ stats) {
    __shared__ __attribute__((aligned(16))) _Float16 tA[TILEM][C0P];
    __shared__ __attribute__((aligned(16))) _Float16 tW[64][C0P];
    __shared__ float sSum[64], sSq[64];

    const int t = threadIdx.x;
    const long long blockRow = (long long)blockIdx.x * TILEM;

    // async-stage pre-converted f16 weights: 64*96*2 B = 768 x 16B chunks
    {
        const unsigned ldsW = lds_addr_of(&tW[0][0]);
        const char* src = (const char*)w0h;
#pragma unroll
        for (int i = 0; i < 3; ++i) {
            int chunk = t + i * 256;
            async_copy_b128(ldsW + chunk * 16, src + chunk * 16);
        }
    }
    if (t < 64) { sSum[t] = 0.f; sSq[t] = 0.f; }

    // stage A tile: gather + center + concat + pad (overlaps async weight fetch)
    {
        int r = t & 127;
        long long g = blockRow + r;
        int b = (int)(g / (SS * KK));
        int rem = (int)(g - (long long)b * (SS * KK));
        int s = rem / KK;
        int k = rem - s * KK;
        int pid = idx[((long long)b * SS + s) * KK + k];
        if (t < 128) {
            const float4* src = (const float4*)(points + ((long long)b * NPT + pid) * CPTS);
#pragma unroll
            for (int c4 = 0; c4 < CPTS / 4; ++c4) {
                float4 v = src[c4];
                tA[r][3 + c4 * 4 + 0] = (_Float16)v.x;
                tA[r][3 + c4 * 4 + 1] = (_Float16)v.y;
                tA[r][3 + c4 * 4 + 2] = (_Float16)v.z;
                tA[r][3 + c4 * 4 + 3] = (_Float16)v.w;
            }
        } else {
            const float* ctr = new_xyz + ((long long)b * SS + s) * 3;
            const float* p = xyz + ((long long)b * NPT + pid) * 3;
            tA[r][0] = (_Float16)(p[0] - ctr[0]);
            tA[r][1] = (_Float16)(p[1] - ctr[1]);
            tA[r][2] = (_Float16)(p[2] - ctr[2]);
            for (int cc = C0; cc < C0P; ++cc) tA[r][cc] = (_Float16)0.f;
        }
    }
    wait_async_zero();   // this wave's async LDS writes done
    __syncthreads();     // publish all waves' LDS writes

    const int wvid = t >> 5, lane = t & 31;
    const int hi4 = lane >> 4, nloc = lane & 15;
    const _Float16* arow = tA[wvid * 16 + nloc];

    f32x8 acc[4] = {};
#pragma unroll
    for (int kb = 0; kb < C0P; kb += 32) {
        f16x16 a = load_frag_a(arow, kb, hi4);
#pragma unroll
        for (int nt = 0; nt < 4; ++nt) {
            f16x16 bf = load_frag_b(tW[nt * 16 + nloc], kb, hi4);
            acc[nt] = __builtin_amdgcn_wmma_f32_16x16x32_f16(
                false, a, false, bf, (short)0, acc[nt], false, false);
        }
    }

    // epilogue: +bias, store f16, accumulate BN stats
#pragma unroll
    for (int nt = 0; nt < 4; ++nt) {
        int n = nt * 16 + nloc;
        float bias = b0[n];
        float psum = 0.f, psq = 0.f;
#pragma unroll
        for (int v = 0; v < 8; ++v) {
            float val = acc[nt][v] + bias;
            long long m = blockRow + wvid * 16 + hi4 * 8 + v;
            x1[m * 64 + n] = (_Float16)val;
            psum += val; psq += val * val;
        }
        atomicAdd(&sSum[n], psum);
        atomicAdd(&sSq[n], psq);
    }
    __syncthreads();
    if (t < 64) {
        atomicAdd(&stats[t], sSum[t]);
        atomicAdd(&stats[128 + t], sSq[t]);
    }
}

// ---------------- 4) finalize BN: scale/shift from sum/sumsq ----------------
__global__ void finalize_stats_kernel(float* stats, const float* __restrict__ g,
                                      const float* __restrict__ be, int C) {
    int t = threadIdx.x;
    if (t < C) {
        float inv = 1.0f / (float)MTOT;
        float mean = stats[t] * inv;
        float var = stats[128 + t] * inv - mean * mean;
        float sc = g[t] * rsqrtf(var + 1e-5f);
        stats[256 + t] = sc;
        stats[384 + t] = be[t] - mean * sc;
    }
}

// ---------------- 5) mid layers: BN+ReLU(prev) -> GEMM(64->NOUT) + stats ----
template <int NOUT, typename OUTT>
__global__ void __launch_bounds__(256) gemm_mid_kernel(const _Float16* __restrict__ xin,
                                                       const _Float16* __restrict__ wh,
                                                       const float* __restrict__ bias,
                                                       const float* __restrict__ bnScale,
                                                       const float* __restrict__ bnShift,
                                                       OUTT* __restrict__ xout,
                                                       float* __restrict__ stats) {
    constexpr int CIN = 64;
    constexpr int NT = NOUT / 16;
    constexpr int WCHUNKS = NOUT * CIN * 2 / 16;   // 16B chunks of the weight tile
    __shared__ __attribute__((aligned(16))) _Float16 tA[TILEM][CIN];
    __shared__ __attribute__((aligned(16))) _Float16 tW[NOUT][CIN];
    __shared__ float sSum[NOUT], sSq[NOUT];

    const int t = threadIdx.x;
    const long long blockRow = (long long)blockIdx.x * TILEM;

    // async-stage pre-converted f16 weights (uniform trip count: EXEC all 1s)
    {
        const unsigned ldsW = lds_addr_of(&tW[0][0]);
        const char* src = (const char*)wh;
#pragma unroll
        for (int i = 0; i < WCHUNKS / 256; ++i) {
            int chunk = t + i * 256;
            async_copy_b128(ldsW + chunk * 16, src + chunk * 16);
        }
    }
    if (t < NOUT) { sSum[t] = 0.f; sSq[t] = 0.f; }

    // stage A: apply BN affine + ReLU of previous layer, f16 into LDS
    for (int i = t; i < TILEM * (CIN / 8); i += 256) {
        int r = i >> 3;
        int cb = (i & 7) * 8;
        h8 v = *(const h8*)(xin + (blockRow + r) * CIN + cb);
        h8 o;
#pragma unroll
        for (int j = 0; j < 8; ++j) {
            float xv = (float)v[j] * bnScale[cb + j] + bnShift[cb + j];
            o[j] = (_Float16)fmaxf(xv, 0.f);
        }
        *(h8*)(&tA[r][cb]) = o;
    }
    wait_async_zero();
    __syncthreads();

    const int wvid = t >> 5, lane = t & 31;
    const int hi4 = lane >> 4, nloc = lane & 15;
    const _Float16* arow = tA[wvid * 16 + nloc];

    f32x8 acc[NT] = {};
#pragma unroll
    for (int kb = 0; kb < CIN; kb += 32) {
        f16x16 a = load_frag_a(arow, kb, hi4);
#pragma unroll
        for (int nt = 0; nt < NT; ++nt) {
            f16x16 bf = load_frag_b(tW[nt * 16 + nloc], kb, hi4);
            acc[nt] = __builtin_amdgcn_wmma_f32_16x16x32_f16(
                false, a, false, bf, (short)0, acc[nt], false, false);
        }
    }

#pragma unroll
    for (int nt = 0; nt < NT; ++nt) {
        int n = nt * 16 + nloc;
        float bs = bias[n];
        float psum = 0.f, psq = 0.f;
#pragma unroll
        for (int v = 0; v < 8; ++v) {
            float val = acc[nt][v] + bs;
            long long m = blockRow + wvid * 16 + hi4 * 8 + v;
            xout[m * NOUT + n] = (OUTT)val;
            psum += val; psq += val * val;
        }
        atomicAdd(&sSum[n], psum);
        atomicAdd(&sSq[n], psq);
    }
    __syncthreads();
    if (t < NOUT) {
        atomicAdd(&stats[t], sSum[t]);
        atomicAdd(&stats[128 + t], sSq[t]);
    }
}

// ---------------- 6) BN+ReLU(layer2) + max-pool over K ----------------
__global__ void __launch_bounds__(256) pool_kernel(const float* __restrict__ x3,
                                                   const float* __restrict__ stats2,
                                                   float* __restrict__ out) {
    const long long id = (long long)blockIdx.x * blockDim.x + threadIdx.x;
    if (id >= (long long)BB * SS * 128) return;
    const long long bs = id >> 7;
    const int ch = (int)(id & 127);
    const float sc = stats2[256 + ch];
    const float sh = stats2[384 + ch];
    float m = 0.f;  // ReLU output is >= 0
#pragma unroll 4
    for (int k = 0; k < KK; ++k) {
        float v = x3[(bs * KK + k) * 128 + ch];
        v = fmaxf(v * sc + sh, 0.f);
        m = fmaxf(m, v);
    }
    out[(long long)BB * SS * 3 + bs * 128 + ch] = m;
}

// ---------------- launch ----------------
extern "C" void kernel_launch(void* const* d_in, const int* in_sizes, int n_in,
                              void* d_out, int out_size, void* d_ws, size_t ws_size,
                              hipStream_t stream) {
    const float* xyz    = (const float*)d_in[0];
    const float* points = (const float*)d_in[1];
    const float* w0 = (const float*)d_in[2];
    const float* b0 = (const float*)d_in[3];
    const float* g0 = (const float*)d_in[4];
    const float* be0 = (const float*)d_in[5];
    const float* w1 = (const float*)d_in[6];
    const float* b1 = (const float*)d_in[7];
    const float* g1 = (const float*)d_in[8];
    const float* be1 = (const float*)d_in[9];
    const float* w2 = (const float*)d_in[10];
    const float* b2 = (const float*)d_in[11];
    const float* g2 = (const float*)d_in[12];
    const float* be2 = (const float*)d_in[13];

    float* out = (float*)d_out;
    float* new_xyz = out;                      // [B,S,3]
    char* ws = (char*)d_ws;

    // workspace layout
    int*      fps_idx = (int*)(ws + 0);                         //  64 KB
    int*      idx     = (int*)(ws + (1ll << 20));               //   2 MB
    float*    stats   = (float*)(ws + (4ll << 20));             //   6 KB (3 x 512 f32)
    _Float16* w0h     = (_Float16*)(ws + (5ll << 20));          //  12 KB [64][96]
    _Float16* w1h     = (_Float16*)(ws + (5ll << 20) + 65536);  //   8 KB [64][64]
    _Float16* w2h     = (_Float16*)(ws + (5ll << 20) + 131072); //  16 KB [128][64]
    _Float16* x1      = (_Float16*)(ws + (8ll << 20));          //  64 MB
    _Float16* x2      = (_Float16*)(ws + (80ll << 20));         //  64 MB
    float*    x3      = (float*)(ws + (152ll << 20));           // 256 MB

    float* stats0 = stats;
    float* stats1 = stats + STATS_STRIDE;
    float* stats2 = stats + 2 * STATS_STRIDE;

    zero_stats_kernel<<<1, 256, 0, stream>>>(stats);
    prep_weights_kernel<<<32, 256, 0, stream>>>(w0, w1, w2, w0h, w1h, w2h);

    fps_kernel<<<BB, 1024, 0, stream>>>(xyz, fps_idx, new_xyz);

    ballq_kernel<<<(BB * SS * 32) / 256, 256, 0, stream>>>(xyz, new_xyz, idx);

    const int nblk = (int)(MTOT / TILEM);  // 4096
    gemm0_kernel<<<nblk, 256, 0, stream>>>(xyz, points, new_xyz, idx, w0h, b0, x1, stats0);
    finalize_stats_kernel<<<1, 128, 0, stream>>>(stats0, g0, be0, 64);

    gemm_mid_kernel<64, _Float16><<<nblk, 256, 0, stream>>>(
        x1, w1h, b1, stats0 + 256, stats0 + 384, x2, stats1);
    finalize_stats_kernel<<<1, 128, 0, stream>>>(stats1, g1, be1, 64);

    gemm_mid_kernel<128, float><<<nblk, 256, 0, stream>>>(
        x2, w2h, b2, stats1 + 256, stats1 + 384, x3, stats2);
    finalize_stats_kernel<<<1, 128, 0, stream>>>(stats2, g2, be2, 128);

    pool_kernel<<<(int)((BB * (long long)SS * 128) / 256), 256, 0, stream>>>(
        x3, stats2, out);
}
